// AutoformerEncoderLayer_45140106281305
// MI455X (gfx1250) — matrix operand order repaired and checked by reading, not compiled, not run on hardware
//
// Autoformer encoder layer for MI455X (gfx1250, wave32, WMMA bf16).
// Direct WMMA autocorrelation instead of FFT: ~36 GFLOP of v_wmma_f32_16x16x32_bf16
// is cheaper than the kernel's ~1.5GB HBM traffic on this chip.
// Round 3: fix async-to-LDS builtin pointer types (int4 vector pointee, AS1 src / AS3 dst).
#include <hip/hip_runtime.h>
#include <hip/hip_bf16.h>
#include <math.h>
#include <stdint.h>

typedef __bf16 bf16_t;
typedef __attribute__((ext_vector_type(16))) __bf16 v16bf;
typedef __attribute__((ext_vector_type(8)))  float  v8f;
typedef __attribute__((ext_vector_type(4)))  float  f4;

#define LDT 40        // LDS row pitch in bf16 elems (80B): 16B aligned, conflict-free frag reads
#define LBUF (128 * LDT)

#if __has_builtin(__builtin_amdgcn_global_load_async_to_lds_b128)
#define HAVE_ASYNC_LDS 1
#endif

// Pointer types matching the builtin signature: pointee = int __vector_size__(16)
typedef int v4i_vs __attribute__((vector_size(16)));
typedef v4i_vs __attribute__((address_space(1))) * gptr128_t;  // global src
typedef v4i_vs __attribute__((address_space(3))) * lptr128_t;  // LDS dst

__device__ __forceinline__ void wait_async_lds(){
#if __has_builtin(__builtin_amdgcn_s_wait_asynccnt)
  __builtin_amdgcn_s_wait_asynccnt(0);
#elif defined(HAVE_ASYNC_LDS)
  asm volatile("s_wait_asynccnt 0" ::: "memory");
#endif
}

// Stage 32B (16 bf16) from global to LDS for this thread.
__device__ __forceinline__ void stage32B(const bf16_t* g, bf16_t* l){
#if defined(HAVE_ASYNC_LDS)
  __builtin_amdgcn_global_load_async_to_lds_b128(
      (gptr128_t)(uintptr_t)g,       (lptr128_t)(unsigned)(uintptr_t)l,       0, 0);
  __builtin_amdgcn_global_load_async_to_lds_b128(
      (gptr128_t)(uintptr_t)(g + 8), (lptr128_t)(unsigned)(uintptr_t)(l + 8), 0, 0);
#else
  f4 a0 = *(const f4*)g; f4 a1 = *(const f4*)(g + 8);
  __builtin_prefetch(g + 32, 0, 3);
  *(f4*)l = a0; *(f4*)(l + 8) = a1;
#endif
}

__device__ __forceinline__ bf16_t to_bf16(float x){
  unsigned u = __float_as_uint(x);
  unsigned r = (u + 0x7FFFu + ((u >> 16) & 1u)) >> 16;   // round-to-nearest-even
  unsigned short h = (unsigned short)r;
  return __builtin_bit_cast(bf16_t, h);
}

// A fragment (16x32 bf16): lane l<16 -> row M=l, K = 0..7 & 16..23; lane>=16 -> K = 8..15 & 24..31
__device__ __forceinline__ v16bf load_afrag(const bf16_t* As, int mbase, int lane){
  const int row = mbase + (lane & 15);
  const int kb  = (lane >> 4) << 3;                 // 0 or 8
  union { v16bf v; f4 f[2]; } u;
  u.f[0] = *(const f4*)(As + row * LDT + kb);
  u.f[1] = *(const f4*)(As + row * LDT + 16 + kb);
  return u.v;
}

// B fragment (32x16 bf16): lane l<16 -> col N=l, K = 0..7 & 16..23; lane>=16 -> K = 8..15 & 24..31 (B staged as [n][k])
__device__ __forceinline__ v16bf load_bfrag(const bf16_t* Bs, int nbase, int lane){
  const int n  = nbase + (lane & 15);
  const int kh = (lane >> 4) << 3;                  // 0 or 8
  union { v16bf v; f4 f[2]; } u;
  u.f[0] = *(const f4*)(Bs + n * LDT + kh);
  u.f[1] = *(const f4*)(Bs + n * LDT + 16 + kh);
  return u.v;
}

// ---------------------------------------------------------------------------
// Moving-average-25 decomposition with edge padding; emits seasonal (f32+bf16)
// and trend (optionally + trend_add). One thread per (b,d) over a 128-t chunk.
// ---------------------------------------------------------------------------
__global__ void decomp_ma25(const float* __restrict__ in, float* __restrict__ s_f32,
                            bf16_t* __restrict__ s_bf16, float* __restrict__ trend_out,
                            const float* __restrict__ trend_add, int T, int D){
  const int d  = blockIdx.x * 256 + threadIdx.x;
  const int b  = blockIdx.z;
  const int t0 = blockIdx.y * 128;
  const float* col = in + (size_t)b * T * D + d;
  float sum = 0.0f;
  for (int s = t0 - 12; s <= t0 + 12; ++s){
    int sc = s < 0 ? 0 : (s > T - 1 ? T - 1 : s);
    sum += col[(size_t)sc * D];
  }
  for (int t = t0; t < t0 + 128; ++t){
    const float trend = sum * (1.0f / 25.0f);
    const float se = col[(size_t)t * D] - trend;
    const size_t off = ((size_t)b * T + t) * D + d;
    s_f32[off]  = se;
    s_bf16[off] = to_bf16(se);
    trend_out[off] = trend + (trend_add ? trend_add[off] : 0.0f);
    const int sa = (t + 13 > T - 1) ? T - 1 : t + 13;
    const int sr = (t - 12 < 0) ? 0 : t - 12;
    sum += col[(size_t)sa * D] - col[(size_t)sr * D];
  }
}

// Convert weight W[K][N] (f32) -> Wt[N][K] (bf16) so GEMM B-tiles are contiguous.
__global__ void weight_prep(const float* __restrict__ W, bf16_t* __restrict__ Wt, int K, int N){
  const int idx = blockIdx.x * 256 + threadIdx.x;
  if (idx >= N * K) return;
  const int n = idx / K, kq = idx % K;
  Wt[idx] = to_bf16(W[(size_t)kq * N + n]);
}

// ---------------------------------------------------------------------------
// WMMA GEMM: C[M][N] = A[M][K](bf16 row-major) x Wt[N][K](bf16) + bias, fp32 acc.
// Block tile 128x128x32; 8 waves of 64x32 (4x2 wmma tiles each).
// Double-buffered LDS: one barrier per K-step; async staging of the next tile.
// MODE: 0 = store bf16; 1 = store f32; 2 = exact GELU -> bf16; 3 = residual + f32
// ---------------------------------------------------------------------------
template<int MODE>
__launch_bounds__(256)
__global__ void gemm_bf16_wmma(const bf16_t* __restrict__ A, const bf16_t* __restrict__ Bt,
                               const float* __restrict__ bias, const float* __restrict__ resid,
                               float* __restrict__ outF, bf16_t* __restrict__ outH,
                               int N, int K){
  __shared__ __align__(16) bf16_t As[2 * LBUF];
  __shared__ __align__(16) bf16_t Bs[2 * LBUF];
  const int tid  = threadIdx.x;
  const int lane = tid & 31, w = tid >> 5;
  const int wm = (w >> 2) * 64, wn = (w & 3) * 32;
  const size_t m0 = (size_t)blockIdx.x * 128;
  const int    n0 = blockIdx.y * 128;
  const int lr = tid >> 1;                 // staged row 0..127
  const int lc = (tid & 1) * 16;           // 0 or 16 (bf16 elems, 32B per thread)
  const bf16_t* ga = A  + (m0 + lr) * (size_t)K + lc;
  const bf16_t* gb = Bt + (size_t)(n0 + lr) * K + lc;
  bf16_t* sa = As + lr * LDT + lc;
  bf16_t* sb = Bs + lr * LDT + lc;

  // Prologue: stage tile 0 into buffer 0
  stage32B(ga, sa); stage32B(gb, sb);
  ga += 32; gb += 32;

  v8f acc[4][2] = {};
  int cur = 0;
  for (int kk = 0; kk < K; kk += 32){
    wait_async_lds();
    __syncthreads();                       // tile `cur` visible to all waves
    if (kk + 32 < K){                      // stage next tile into other buffer
      stage32B(ga, sa + (cur ^ 1) * LBUF);
      stage32B(gb, sb + (cur ^ 1) * LBUF);
      ga += 32; gb += 32;
    }
    const bf16_t* Ab = As + cur * LBUF;
    const bf16_t* Bb = Bs + cur * LBUF;
    v16bf bf0 = load_bfrag(Bb, wn,      lane);
    v16bf bf1 = load_bfrag(Bb, wn + 16, lane);
#pragma unroll
    for (int i = 0; i < 4; ++i){
      v16bf af = load_afrag(Ab, wm + i * 16, lane);
      acc[i][0] = __builtin_amdgcn_wmma_f32_16x16x32_bf16(false, af, false, bf0, (short)0, acc[i][0], false, false);
      acc[i][1] = __builtin_amdgcn_wmma_f32_16x16x32_bf16(false, af, false, bf1, (short)0, acc[i][1], false, false);
    }
    cur ^= 1;
  }
  // C layout: lane<16 -> N=lane, M=r; lane>=16 -> N=lane-16, M=r+8
  const int lhi = lane >> 4, ln = lane & 15;
#pragma unroll
  for (int i = 0; i < 4; ++i){
#pragma unroll
    for (int j = 0; j < 2; ++j){
      const int ncol = n0 + wn + j * 16 + ln;
      const float bv = bias[ncol];
      const size_t mbase = m0 + wm + i * 16 + lhi * 8;
#pragma unroll
      for (int r = 0; r < 8; ++r){
        const size_t off = (mbase + r) * (size_t)N + ncol;
        float c = acc[i][j][r] + bv;
        if (MODE == 0)      outH[off] = to_bf16(c);
        else if (MODE == 1) outF[off] = c;
        else if (MODE == 2){ float g = 0.5f * c * (1.0f + erff(c * 0.70710678118654752f)); outH[off] = to_bf16(g); }
        else                outF[off] = resid[off] + c;
      }
    }
  }
}

// ---------------------------------------------------------------------------
// Autocorrelation score: lower-triangular 128x128 blocks of G = Q K^T per batch.
// score[b,tau] = sum over diagonal tau. Each block writes 255 diagonal partials
// to its private slot partial[b][bi-bj][bi][0..254]  -> fully deterministic.
// ---------------------------------------------------------------------------
__launch_bounds__(256)
__global__ void autocorr_score_wmma(const bf16_t* __restrict__ q, const bf16_t* __restrict__ k,
                                    float* __restrict__ partial, int T, int D){
  __shared__ __align__(16) bf16_t As[2 * LBUF];
  __shared__ __align__(16) bf16_t Bs[2 * LBUF];
  __shared__ float part[16 * 256];   // per (wave, lane-half) private diagonal slices
  const int b = blockIdx.y;
  const int p = blockIdx.x;
  int bi = 0;
  while (((bi + 1) * (bi + 2)) / 2 <= p) ++bi;     // lower-tri block enumeration
  const int bj = p - (bi * (bi + 1)) / 2;

  const int tid  = threadIdx.x;
  const int lane = tid & 31, w = tid >> 5;
  const int wm = (w >> 2) * 64, wn = (w & 3) * 32;
  const int lr = tid >> 1, lc = (tid & 1) * 16;
  const bf16_t* ga = q + ((size_t)b * T + bi * 128 + lr) * D + lc;
  const bf16_t* gb = k + ((size_t)b * T + bj * 128 + lr) * D + lc;
  bf16_t* sa = As + lr * LDT + lc;
  bf16_t* sb = Bs + lr * LDT + lc;

  for (int i = tid; i < 16 * 256; i += 256) part[i] = 0.0f;

  stage32B(ga, sa); stage32B(gb, sb);
  ga += 32; gb += 32;

  v8f acc[4][2] = {};
  int cur = 0;
  for (int kk = 0; kk < D; kk += 32){
    wait_async_lds();
    __syncthreads();
    if (kk + 32 < D){
      stage32B(ga, sa + (cur ^ 1) * LBUF);
      stage32B(gb, sb + (cur ^ 1) * LBUF);
      ga += 32; gb += 32;
    }
    const bf16_t* Ab = As + cur * LBUF;
    const bf16_t* Bb = Bs + cur * LBUF;
    v16bf bf0 = load_bfrag(Bb, wn,      lane);
    v16bf bf1 = load_bfrag(Bb, wn + 16, lane);
#pragma unroll
    for (int i = 0; i < 4; ++i){
      v16bf af = load_afrag(Ab, wm + i * 16, lane);
      acc[i][0] = __builtin_amdgcn_wmma_f32_16x16x32_bf16(false, af, false, bf0, (short)0, acc[i][0], false, false);
      acc[i][1] = __builtin_amdgcn_wmma_f32_16x16x32_bf16(false, af, false, bf1, (short)0, acc[i][1], false, false);
    }
    cur ^= 1;
  }
  // Diagonal accumulation. tau = (bi-bj)*128 + (M - N); local = (M-N)+127 in [0,254].
  // Within one vector op all 16 lanes of a half-wave slice hit distinct addresses.
  const int lhi = lane >> 4, ln = lane & 15;
  float* slice = part + (w * 2 + lhi) * 256;
#pragma unroll
  for (int i = 0; i < 4; ++i){
#pragma unroll
    for (int j = 0; j < 2; ++j){
      const int Mb = wm + i * 16 + lhi * 8;
      const int Nc = wn + j * 16 + ln;
#pragma unroll
      for (int r = 0; r < 8; ++r)
        slice[Mb + r - Nc + 127] += acc[i][j][r];
    }
  }
  __syncthreads();
  if (tid < 255){
    float s = 0.0f;
#pragma unroll
    for (int sl = 0; sl < 16; ++sl) s += part[sl * 256 + tid];
    partial[(((size_t)b * 16 + (bi - bj)) * 16 + bi) * 256 + tid] = s;
  }
}

// Per-batch: reduce block partials into score[0..T), mask tau=0, pick top-8 lags.
__global__ void score_reduce_topk(const float* __restrict__ partial, int* __restrict__ lags,
                                  int T, int K8){
  const int b = blockIdx.x;
  __shared__ float s[2048];
  __shared__ float rmax[256];
  __shared__ int   rarg[256];
  const int tid = threadIdx.x;
  for (int tau = tid; tau < T; tau += 256){
    float acc = 0.0f;
    const int g_lo = (tau >= 128) ? (tau >> 7) : 0;
    int g_hi = (tau + 127) >> 7; if (g_hi > 15) g_hi = 15;
    for (int g = g_lo; g <= g_hi; ++g){
      const int local = tau - g * 128 + 127;                  // 0..254
      const float* base = partial + (((size_t)b * 16 + g) * 16) * 256 + local;
      for (int biq = g; biq < 16; ++biq) acc += base[(size_t)biq * 256];
    }
    s[tau] = acc;
  }
  __syncthreads();
  if (tid == 0) s[0] = -1e30f;
  __syncthreads();
  for (int kq = 0; kq < K8; ++kq){
    float best = -1e38f; int bidx = 0;
    for (int i = tid; i < T; i += 256){ if (s[i] > best){ best = s[i]; bidx = i; } }
    rmax[tid] = best; rarg[tid] = bidx;
    __syncthreads();
    for (int off = 128; off > 0; off >>= 1){
      if (tid < off && rmax[tid + off] > rmax[tid]){ rmax[tid] = rmax[tid + off]; rarg[tid] = rarg[tid + off]; }
      __syncthreads();
    }
    if (tid == 0){ lags[b * K8 + kq] = rarg[0]; s[rarg[0]] = -1e38f; }
    __syncthreads();
  }
}

// s[b,t,:] += (1/K) * sum_k v[b, (t - lag_k) mod T, :]   (in place)
__global__ void aggregate_shift(float* __restrict__ s, const float* __restrict__ v,
                                const int* __restrict__ lags, int T, int D, int K8){
  const int t = blockIdx.x, b = blockIdx.y;
  const int d = threadIdx.x * 4;
  f4 acc = {0.f, 0.f, 0.f, 0.f};
  for (int kq = 0; kq < K8; ++kq){
    int lag = lags[b * K8 + kq];
    int idx = t - lag; if (idx < 0) idx += T;
    acc += *(const f4*)(v + ((size_t)b * T + idx) * D + d);
  }
  const size_t off = ((size_t)b * T + t) * D + d;
  f4 cur = *(const f4*)(s + off);
  cur += acc * (1.0f / (float)K8);
  *(f4*)(s + off) = cur;
}

extern "C" void kernel_launch(void* const* d_in, const int* in_sizes, int n_in,
                              void* d_out, int out_size, void* d_ws, size_t ws_size,
                              hipStream_t stream){
  (void)in_sizes; (void)n_in; (void)out_size; (void)ws_size;
  const float* x  = (const float*)d_in[0];
  const float* Wq = (const float*)d_in[1];
  const float* bq = (const float*)d_in[2];
  const float* Wk = (const float*)d_in[3];
  const float* bk = (const float*)d_in[4];
  const float* Wv = (const float*)d_in[5];
  const float* bv = (const float*)d_in[6];
  const float* W1 = (const float*)d_in[7];
  const float* b1 = (const float*)d_in[8];
  const float* W2 = (const float*)d_in[9];
  const float* b2 = (const float*)d_in[10];

  const int B = 16, T = 2048, D = 512, F = 2048, K8 = 8;
  const size_t BTD = (size_t)B * T * D;          // 16.78M elems
  float* seasonal_out = (float*)d_out;           // [B,T,D] f32
  float* trend_out    = (float*)d_out + BTD;     // [B,T,D] f32

  unsigned char* wsb = (unsigned char*)d_ws;
  size_t woff = 0;
  auto take = [&](size_t bytes) -> void* {
    void* p = wsb + woff; woff += (bytes + 255) & ~(size_t)255; return p;
  };
  bf16_t* s1h  = (bf16_t*)take(BTD * 2);                     // seasonal1 bf16
  bf16_t* qh   = (bf16_t*)take(BTD * 2);
  bf16_t* khb  = (bf16_t*)take(BTD * 2);
  float*  vf   = (float*) take(BTD * 4);
  float*  s1f  = (float*) take(BTD * 4);                     // seasonal1 f32 (later s_mid in place)
  float*  t1f  = (float*) take(BTD * 4);                     // trend1
  float*  s2f  = (float*) take(BTD * 4);
  bf16_t* s2h  = (bf16_t*)take(BTD * 2);
  bf16_t* hh   = (bf16_t*)take((size_t)B * T * F * 2);       // GELU(FFN1) bf16
  float*  part = (float*) take((size_t)B * 16 * 16 * 256 * 4);
  int*    lags = (int*)   take((size_t)B * K8 * 4);
  bf16_t* WqT  = (bf16_t*)take((size_t)D * D * 2);
  bf16_t* WkT  = (bf16_t*)take((size_t)D * D * 2);
  bf16_t* WvT  = (bf16_t*)take((size_t)D * D * 2);
  bf16_t* W1T  = (bf16_t*)take((size_t)D * F * 2);
  bf16_t* W2T  = (bf16_t*)take((size_t)F * D * 2);

  // Weight transpose+convert to bf16 [N][K]
  weight_prep<<<(D * D + 255) / 256, 256, 0, stream>>>(Wq, WqT, D, D);
  weight_prep<<<(D * D + 255) / 256, 256, 0, stream>>>(Wk, WkT, D, D);
  weight_prep<<<(D * D + 255) / 256, 256, 0, stream>>>(Wv, WvT, D, D);
  weight_prep<<<(D * F + 255) / 256, 256, 0, stream>>>(W1, W1T, D, F);
  weight_prep<<<(F * D + 255) / 256, 256, 0, stream>>>(W2, W2T, F, D);

  // Decomp 1: x -> seasonal1 (f32 + bf16), trend1
  dim3 gd(D / 256, T / 128, B);
  decomp_ma25<<<gd, 256, 0, stream>>>(x, s1f, s1h, t1f, nullptr, T, D);

  // Q/K/V projections (M = 32768)
  dim3 gqkv(256, D / 128);
  gemm_bf16_wmma<0><<<gqkv, 256, 0, stream>>>(s1h, WqT, bq, nullptr, nullptr, qh,  D, D);
  gemm_bf16_wmma<0><<<gqkv, 256, 0, stream>>>(s1h, WkT, bk, nullptr, nullptr, khb, D, D);
  gemm_bf16_wmma<1><<<gqkv, 256, 0, stream>>>(s1h, WvT, bv, nullptr, vf,      nullptr, D, D);

  // Correlation score partials over lower-triangular 128x128 blocks, then top-8
  dim3 gsc(136, B);
  autocorr_score_wmma<<<gsc, 256, 0, stream>>>(qh, khb, part, T, D);
  score_reduce_topk<<<B, 256, 0, stream>>>(part, lags, T, K8);

  // seasonal1 += mean of 8 shifted v  (in place on s1f)
  dim3 gag(T, B);
  aggregate_shift<<<gag, 128, 0, stream>>>(s1f, vf, lags, T, D, K8);

  // Decomp 2: -> seasonal2 (f32 + bf16), trend_out = trend1 + trend2
  decomp_ma25<<<gd, 256, 0, stream>>>(s1f, s2f, s2h, trend_out, t1f, T, D);

  // FFN: h = gelu(s2 @ W1 + b1);  seasonal_out = s2 + h @ W2 + b2
  dim3 g1(256, F / 128);
  gemm_bf16_wmma<2><<<g1, 256, 0, stream>>>(s2h, W1T, b1, nullptr, nullptr, hh, F, D);
  dim3 g2(256, D / 128);
  gemm_bf16_wmma<3><<<g2, 256, 0, stream>>>(hh, W2T, b2, s2f, seasonal_out, nullptr, D, F);
}
